// EncoderLayer_21492016349906
// MI455X (gfx1250) — matrix operand order repaired and checked by reading, not compiled, not run on hardware
//
#include <hip/hip_runtime.h>
#include <math.h>

typedef __attribute__((ext_vector_type(16))) _Float16 v16h;
typedef __attribute__((ext_vector_type(8)))  _Float16 v8h;
typedef __attribute__((ext_vector_type(8)))  float    v8f;

#define B_    16
#define T_    512
#define D_    512
#define H_    8
#define DH_   64
#define COUT_ 7
#define PRED_ 96
#define K_    16
#define DFF_  2048
#define TP_   (T_ + PRED_)          /* 608 */
#define TWO_PI 6.28318530717958647692f

// ---------------------------------------------------------------------------
// Generic f32 -> f16 conversion
// ---------------------------------------------------------------------------
__global__ void f2h_kernel(const float* __restrict__ in, _Float16* __restrict__ out, int n)
{
    int i = blockIdx.x * blockDim.x + threadIdx.x;
    if (i < n) out[i] = (_Float16)in[i];
}

// ---------------------------------------------------------------------------
// DFT basis as "weight" matrix (N=512 rows: 256 Re then 256 Im, K=512 = t)
// W[n][t] : n<256 -> cos(2*pi*(n+1)*t/512) ; n>=256 -> -sin(2*pi*(n-255)*t/512)
// ---------------------------------------------------------------------------
__global__ void basis_kernel(_Float16* __restrict__ Wdft)
{
    int i = blockIdx.x * blockDim.x + threadIdx.x;   // n*512 + t
    if (i >= 512 * 512) return;
    int t = i & 511, n = i >> 9;
    int f = (n & 255) + 1;                 // freq bin 1..256
    int m = (f * t) & 511;                 // exact argument reduction (period 512)
    float ang = (float)m * (TWO_PI / 512.0f);
    float v = (n < 256) ? __cosf(ang) : -__sinf(ang);
    Wdft[i] = (_Float16)v;
}

// ---------------------------------------------------------------------------
// Transpose res (B,T,D) -> resT (B*D rows, T cols) in f16
// ---------------------------------------------------------------------------
__global__ void transpose_h_kernel(const float* __restrict__ res, _Float16* __restrict__ resT)
{
    size_t i = (size_t)blockIdx.x * blockDim.x + threadIdx.x;  // (b*512+d)*512 + t
    if (i >= (size_t)B_ * D_ * T_) return;
    int t = (int)(i & 511);
    int bd = (int)(i >> 9);
    int d = bd & 511, b = bd >> 9;
    resT[i] = (_Float16)res[((size_t)(b * T_ + t) << 9) + d];
}

// ---------------------------------------------------------------------------
// WMMA GEMM:  C(MxN) = A(MxK, f16 row-major) x W(NxK, f16 row-major)^T [+bias]
// Each wave computes a (16*MT) x (16*NT) output tile: MT x NT accumulators,
// A fragments reused NT times, B fragments reused MT times.
// EPI==0 : float out (+ optional bias) ; EPI==1 : relu -> f16 out
// ---------------------------------------------------------------------------
template<int MT, int NT, int EPI>
__global__ __launch_bounds__(128)
void gemm_wmma_kernel(const _Float16* __restrict__ A, int lda,
                      const _Float16* __restrict__ W, int ldw,
                      const float* __restrict__ bias,
                      void* __restrict__ Cout, int ldc,
                      int M, int N, int K)
{
    const int wv   = threadIdx.x >> 5;
    const int lane = threadIdx.x & 31;
    const int half = lane >> 4;
    const int lr   = lane & 15;
    const int n0 = blockIdx.x * (16 * NT);
    const int m0 = (blockIdx.y * 4 + wv) * (16 * MT);
    if (m0 >= M) return;

    v8f acc[MT][NT];
#pragma unroll
    for (int mi = 0; mi < MT; ++mi)
#pragma unroll
        for (int ni = 0; ni < NT; ++ni)
            acc[mi][ni] = (v8f){};

    const _Float16* Ap[MT];
    const _Float16* Wp[NT];
#pragma unroll
    for (int mi = 0; mi < MT; ++mi)
        Ap[mi] = A + (size_t)(m0 + mi * 16 + lr) * lda + half * 8;
#pragma unroll
    for (int ni = 0; ni < NT; ++ni)
        Wp[ni] = W + (size_t)(n0 + ni * 16 + lr) * ldw + half * 8;

    for (int k0 = 0; k0 < K; k0 += 32) {
        v16h a[MT], b[NT];
#pragma unroll
        for (int mi = 0; mi < MT; ++mi) {
            v8h lo = *(const v8h*)(Ap[mi] + k0);
            v8h hi = *(const v8h*)(Ap[mi] + k0 + 16);
#pragma unroll
            for (int i = 0; i < 8; ++i) { a[mi][i] = lo[i]; a[mi][i + 8] = hi[i]; }
        }
#pragma unroll
        for (int ni = 0; ni < NT; ++ni) {
            v8h lo = *(const v8h*)(Wp[ni] + k0);
            v8h hi = *(const v8h*)(Wp[ni] + k0 + 16);
#pragma unroll
            for (int i = 0; i < 8; ++i) { b[ni][i] = lo[i]; b[ni][i + 8] = hi[i]; }
        }
        if (k0 + 32 < K) {
#pragma unroll
            for (int mi = 0; mi < MT; ++mi) __builtin_prefetch(Ap[mi] + k0 + 32, 0, 1);
#pragma unroll
            for (int ni = 0; ni < NT; ++ni) __builtin_prefetch(Wp[ni] + k0 + 32, 0, 1);
        }
#pragma unroll
        for (int mi = 0; mi < MT; ++mi)
#pragma unroll
            for (int ni = 0; ni < NT; ++ni)
                acc[mi][ni] = __builtin_amdgcn_wmma_f32_16x16x32_f16(
                    false, a[mi], false, b[ni], (short)0, acc[mi][ni], false, false);
    }

#pragma unroll
    for (int ni = 0; ni < NT; ++ni) {
        const int col = n0 + ni * 16 + lr;
        float bb = 0.0f;
        if (EPI == 0 && bias != nullptr) bb = bias[col];
#pragma unroll
        for (int mi = 0; mi < MT; ++mi) {
#pragma unroll
            for (int r = 0; r < 8; ++r) {
                int row = m0 + mi * 16 + half * 8 + r;
                if constexpr (EPI == 0) {
                    ((float*)Cout)[(size_t)row * ldc + col] = acc[mi][ni][r] + bb;
                } else {
                    ((_Float16*)Cout)[(size_t)row * ldc + col] =
                        (_Float16)fmaxf(acc[mi][ni][r], 0.0f);
                }
            }
        }
    }
}

// ---------------------------------------------------------------------------
// Top-K over |X(f)|^2, bins 1..255 (cols 0..254 of spect row; col 255 = Nyquist
// excluded). One wave per (b,d) row.
// ---------------------------------------------------------------------------
__global__ __launch_bounds__(256)
void topk_kernel(const float* __restrict__ spect,
                 float* __restrict__ amp, float* __restrict__ fbin, float* __restrict__ ph)
{
    int row  = (int)((blockIdx.x * blockDim.x + threadIdx.x) >> 5);
    int lane = threadIdx.x & 31;
    if (row >= B_ * D_) return;
    const float* re = spect + (size_t)row * 512;
    const float* im = re + 256;

    float mag[8];
#pragma unroll
    for (int j = 0; j < 8; ++j) {
        int c = lane * 8 + j;
        float r = re[c], i = im[c];
        mag[j] = (c < 255) ? (r * r + i * i) : -1.0f;
    }
    for (int sel = 0; sel < K_; ++sel) {
        float bv = -2.0f; int bj = 0;
#pragma unroll
        for (int j = 0; j < 8; ++j)
            if (mag[j] > bv) { bv = mag[j]; bj = j; }
        int bc = lane * 8 + bj;
        for (int o = 16; o > 0; o >>= 1) {
            float ov = __shfl_xor(bv, o, 32);
            int   oc = __shfl_xor(bc, o, 32);
            if (ov > bv || (ov == bv && oc < bc)) { bv = ov; bc = oc; }
        }
        if (lane == sel) {
            float r = re[bc], i = im[bc];
            amp [row * K_ + sel] = 2.0f * sqrtf(fmaxf(bv, 0.0f)) / (float)T_;
            fbin[row * K_ + sel] = (float)(bc + 1);
            ph  [row * K_ + sel] = atan2f(i, r);
        }
        int lj = bc - lane * 8;
        if (lj >= 0 && lj < 8) mag[lj] = -2.0f;
    }
}

// ---------------------------------------------------------------------------
// season[b,tt,d] = sum_j amp_j * cos(2*pi*bin_j*tt/512 + phi_j), tt in [0,608)
// One block per (b,d).
// ---------------------------------------------------------------------------
__global__ __launch_bounds__(256)
void season_kernel(const float* __restrict__ amp, const float* __restrict__ fbin,
                   const float* __restrict__ ph, float* __restrict__ season)
{
    int row = blockIdx.x;               // b*512 + d
    int b = row >> 9, d = row & 511;
    __shared__ float sa[K_]; __shared__ int sf[K_]; __shared__ float sph[K_];
    if (threadIdx.x < K_) {
        sa [threadIdx.x] = amp [row * K_ + threadIdx.x];
        sf [threadIdx.x] = (int)fbin[row * K_ + threadIdx.x];
        sph[threadIdx.x] = ph  [row * K_ + threadIdx.x];
    }
    __syncthreads();
    for (int tt = threadIdx.x; tt < TP_; tt += blockDim.x) {
        float acc = 0.0f;
#pragma unroll
        for (int j = 0; j < K_; ++j) {
            int m = (sf[j] * tt) & (T_ - 1);        // exact mod-512 reduction
            acc += sa[j] * __cosf((float)m * (TWO_PI / (float)T_) + sph[j]);
        }
        season[((size_t)(b * TP_ + tt) << 9) + d] = acc;
    }
}

// ---------------------------------------------------------------------------
// res1 = res - season[:, :T]  (f32 + f16)
// ---------------------------------------------------------------------------
__global__ void res1_kernel(const float* __restrict__ res, const float* __restrict__ season,
                            float* __restrict__ r32, _Float16* __restrict__ r16)
{
    size_t i = (size_t)blockIdx.x * blockDim.x + threadIdx.x;
    if (i >= (size_t)B_ * T_ * D_) return;
    int d = (int)(i & 511);
    int bt = (int)(i >> 9);
    int t = bt & 511, b = bt >> 9;
    float v = res[i] - season[((size_t)(b * TP_ + t) << 9) + d];
    r32[i] = v;
    r16[i] = (_Float16)v;
}

// ---------------------------------------------------------------------------
// Growth ES scan: diffs of v (with z0 prepended), s_t = a*s_{t-1} + (1-a)*diff_t,
// gr = [v0_g, s_0..s_{T-1}] stored f16 for the Wo GEMM.
// ---------------------------------------------------------------------------
__global__ void es_growth_kernel(const float* __restrict__ v, const float* __restrict__ z0,
                                 const float* __restrict__ v0g, const float* __restrict__ swg,
                                 _Float16* __restrict__ gr)
{
    int ch = blockIdx.x * blockDim.x + threadIdx.x;   // b*512 + (h*64+dh)
    if (ch >= B_ * D_) return;
    int b = ch >> 9, hd = ch & 511, h = hd >> 6;
    float alpha = 1.0f / (1.0f + __expf(-swg[h]));
    float onem  = 1.0f - alpha;
    float vprev = z0[hd];
    float s     = v0g[hd];
    gr[((size_t)(b * 513) << 9) + hd] = (_Float16)s;
    for (int t = 0; t < T_; ++t) {
        float vt = v[((size_t)(b * T_ + t) << 9) + hd];
        s = alpha * s + onem * (vt - vprev);
        vprev = vt;
        gr[((size_t)(b * 513 + t + 1) << 9) + hd] = (_Float16)s;
    }
}

// ---------------------------------------------------------------------------
// LayerNorm helpers (blockDim = 256, D = 512, 2 elems/thread)
// ---------------------------------------------------------------------------
__device__ __forceinline__ float block_sum(float v, float* sh)
{
    int lane = threadIdx.x & 31, wid = threadIdx.x >> 5;
#pragma unroll
    for (int o = 16; o > 0; o >>= 1) v += __shfl_xor(v, o, 32);
    if (lane == 0) sh[wid] = v;
    __syncthreads();
    if (wid == 0) {
        float r = (lane < 8) ? sh[lane] : 0.0f;
#pragma unroll
        for (int o = 4; o > 0; o >>= 1) r += __shfl_xor(r, o, 32);
        if (lane == 0) sh[0] = r;
    }
    __syncthreads();
    float r = sh[0];
    __syncthreads();
    return r;
}

__global__ __launch_bounds__(256)
void ln1_kernel(const float* __restrict__ res1, const float* __restrict__ growth,
                const float* __restrict__ g, const float* __restrict__ beta,
                float* __restrict__ h32, _Float16* __restrict__ h16)
{
    __shared__ float sh[8];
    int row = blockIdx.x;                 // b*512 + t
    int b = row >> 9, t = row & 511;
    const float* xr = res1 + ((size_t)row << 9);
    const float* xg = growth + ((size_t)(b * 513 + t + 1) << 9);
    int d0 = threadIdx.x, d1 = threadIdx.x + 256;
    float x0 = xr[d0] - xg[d0];
    float x1 = xr[d1] - xg[d1];
    float mu  = block_sum(x0 + x1, sh) * (1.0f / 512.0f);
    float e0 = x0 - mu, e1 = x1 - mu;
    float var = block_sum(e0 * e0 + e1 * e1, sh) * (1.0f / 512.0f);
    float inv = rsqrtf(var + 1e-5f);
    float o0 = e0 * inv * g[d0] + beta[d0];
    float o1 = e1 * inv * g[d1] + beta[d1];
    size_t base = (size_t)row << 9;
    h32[base + d0] = o0;  h32[base + d1] = o1;
    h16[base + d0] = (_Float16)o0;  h16[base + d1] = (_Float16)o1;
}

__global__ __launch_bounds__(256)
void ln2_kernel(const float* __restrict__ hbuf, const float* __restrict__ ff,
                const float* __restrict__ g, const float* __restrict__ beta,
                float* __restrict__ out)
{
    __shared__ float sh[8];
    int row = blockIdx.x;
    size_t base = (size_t)row << 9;
    int d0 = threadIdx.x, d1 = threadIdx.x + 256;
    float x0 = hbuf[base + d0] + ff[base + d0];
    float x1 = hbuf[base + d1] + ff[base + d1];
    float mu  = block_sum(x0 + x1, sh) * (1.0f / 512.0f);
    float e0 = x0 - mu, e1 = x1 - mu;
    float var = block_sum(e0 * e0 + e1 * e1, sh) * (1.0f / 512.0f);
    float inv = rsqrtf(var + 1e-5f);
    out[base + d0] = e0 * inv * g[d0] + beta[d0];
    out[base + d1] = e1 * inv * g[d1] + beta[d1];
}

// ---------------------------------------------------------------------------
// N=7 projections (gp / sp): out[b,t,c] = X[row(b,t)] . W[c] + bias[c]
// ---------------------------------------------------------------------------
__global__ void proj7_kernel(const float* __restrict__ X, int rowsPerB, int tOff,
                             const float* __restrict__ W, const float* __restrict__ bias,
                             float* __restrict__ out)
{
    int id = blockIdx.x * blockDim.x + threadIdx.x;
    if (id >= B_ * T_ * COUT_) return;
    int c = id % COUT_;
    int bt = id / COUT_;
    int t = bt & 511, b = bt >> 9;
    const float* x = X + ((size_t)(b * rowsPerB + t + tOff) << 9);
    const float* w = W + c * 512;
    float acc = bias[c];
    for (int d = 0; d < 512; ++d) acc += x[d] * w[d];
    out[(size_t)bt * COUT_ + c] = acc;
}

// ---------------------------------------------------------------------------
// Level ES with aux term: s_t = a*s_{t-1} + (1-a)*(level-sp)_t + a*gp_t
// ---------------------------------------------------------------------------
__global__ void level_es_kernel(const float* __restrict__ level, const float* __restrict__ sp,
                                const float* __restrict__ gp, const float* __restrict__ v0l,
                                const float* __restrict__ swl, float* __restrict__ lvl)
{
    int ch = blockIdx.x * blockDim.x + threadIdx.x;   // b*7 + c
    if (ch >= B_ * COUT_) return;
    int b = ch / COUT_, c = ch % COUT_;
    float alpha = 1.0f / (1.0f + __expf(-swl[c]));
    float onem  = 1.0f - alpha;
    float s = v0l[c];
    for (int t = 0; t < T_; ++t) {
        size_t idx = (size_t)(b * T_ + t) * COUT_ + c;
        s = alpha * s + onem * (level[idx] - sp[idx]) + alpha * gp[idx];
        lvl[idx] = s;
    }
}

// ---------------------------------------------------------------------------
// Host-side orchestration
// ---------------------------------------------------------------------------
extern "C" void kernel_launch(void* const* d_in, const int* in_sizes, int n_in,
                              void* d_out, int out_size, void* d_ws, size_t ws_size,
                              hipStream_t stream)
{
    (void)in_sizes; (void)n_in; (void)out_size; (void)ws_size;

    const float* res   = (const float*)d_in[0];
    const float* level = (const float*)d_in[1];
    const float* Wi    = (const float*)d_in[2];
    const float* bi    = (const float*)d_in[3];
    const float* z0    = (const float*)d_in[4];
    const float* v0g   = (const float*)d_in[5];
    const float* swg   = (const float*)d_in[6];
    const float* Wo    = (const float*)d_in[7];
    const float* bo    = (const float*)d_in[8];
    const float* W1    = (const float*)d_in[9];
    const float* W2    = (const float*)d_in[10];
    const float* g1    = (const float*)d_in[11];
    const float* b1ln  = (const float*)d_in[12];
    const float* g2    = (const float*)d_in[13];
    const float* b2ln  = (const float*)d_in[14];
    const float* Wg    = (const float*)d_in[15];
    const float* bg    = (const float*)d_in[16];
    const float* Ws    = (const float*)d_in[17];
    const float* bs    = (const float*)d_in[18];
    const float* v0l   = (const float*)d_in[19];
    const float* swl   = (const float*)d_in[20];

    float* out        = (float*)d_out;
    float* h_out      = out;                                 // B*T*D
    float* lvl_out    = out + (size_t)B_ * T_ * D_;          // B*T*COUT
    float* growth_out = lvl_out + (size_t)B_ * T_ * COUT_;   // B*(T+1)*D
    float* season_out = growth_out + (size_t)B_ * (T_ + 1) * D_;  // B*TP*D

    // workspace carve-up
    char*  ws  = (char*)d_ws;
    size_t off = 0;
    auto alloc = [&](size_t bytes) -> void* {
        void* p = ws + off;
        off += (bytes + 255) & ~(size_t)255;
        return p;
    };
    const size_t MD  = (size_t)B_ * T_ * D_;       // 4,194,304
    _Float16* resT_h   = (_Float16*)alloc(MD * 2);
    _Float16* Wdft_h   = (_Float16*)alloc((size_t)512 * 512 * 2);
    float*    spect    = (float*)   alloc(MD * 4);            // 8192 x 512 (re|im)
    float*    amp      = (float*)   alloc((size_t)B_ * D_ * K_ * 4);
    float*    fbin     = (float*)   alloc((size_t)B_ * D_ * K_ * 4);
    float*    ph       = (float*)   alloc((size_t)B_ * D_ * K_ * 4);
    _Float16* Wi_h     = (_Float16*)alloc((size_t)512 * 512 * 2);
    _Float16* Wo_h     = (_Float16*)alloc((size_t)512 * 512 * 2);
    _Float16* W1_h     = (_Float16*)alloc((size_t)DFF_ * 512 * 2);
    _Float16* W2_h     = (_Float16*)alloc((size_t)512 * DFF_ * 2);
    float*    res1_32  = (float*)   alloc(MD * 4);
    _Float16* res1_h   = (_Float16*)alloc(MD * 2);
    float*    v_f32    = (float*)   alloc(MD * 4);            // reused later as ff
    _Float16* gr_h     = (_Float16*)alloc((size_t)B_ * 513 * D_ * 2);
    float*    h_f32    = (float*)   alloc(MD * 4);
    _Float16* h_h      = (_Float16*)alloc(MD * 2);
    _Float16* hidden_h = (_Float16*)alloc((size_t)B_ * T_ * DFF_ * 2);
    float*    gp       = (float*)   alloc((size_t)B_ * T_ * COUT_ * 4);
    float*    sp       = (float*)   alloc((size_t)B_ * T_ * COUT_ * 4);

    const dim3 blk256(256), blkG(128);

    // 1. weight conversions + DFT basis
    f2h_kernel<<<dim3(1024), blk256, 0, stream>>>(Wi, Wi_h, 512 * 512);
    f2h_kernel<<<dim3(1024), blk256, 0, stream>>>(Wo, Wo_h, 512 * 512);
    f2h_kernel<<<dim3(4096), blk256, 0, stream>>>(W1, W1_h, DFF_ * 512);
    f2h_kernel<<<dim3(4096), blk256, 0, stream>>>(W2, W2_h, 512 * DFF_);
    basis_kernel<<<dim3(1024), blk256, 0, stream>>>(Wdft_h);

    // 2. transpose res; spectrum GEMM (WMMA); top-k; season reconstruction
    transpose_h_kernel<<<dim3(16384), blk256, 0, stream>>>(res, resT_h);
    // M=8192,N=512,K=512 ; wave tile 32x64 -> block covers 128 rows, 64 cols
    gemm_wmma_kernel<2, 4, 0><<<dim3(512 / 64, 8192 / 128), blkG, 0, stream>>>(
        resT_h, 512, Wdft_h, 512, nullptr, spect, 512, 8192, 512, 512);
    topk_kernel<<<dim3(1024), blk256, 0, stream>>>(spect, amp, fbin, ph);
    season_kernel<<<dim3(8192), blk256, 0, stream>>>(amp, fbin, ph, season_out);

    // 3. res1 = res - season[:, :T]
    res1_kernel<<<dim3(16384), blk256, 0, stream>>>(res, season_out, res1_32, res1_h);

    // 4. v = res1 @ Wi^T + bi  (WMMA)
    gemm_wmma_kernel<2, 4, 0><<<dim3(512 / 64, 8192 / 128), blkG, 0, stream>>>(
        res1_h, 512, Wi_h, 512, bi, v_f32, 512, 8192, 512, 512);

    // 5. growth ES scan -> gr (f16), then growth = gr @ Wo^T + bo (WMMA)
    es_growth_kernel<<<dim3(32), blk256, 0, stream>>>(v_f32, z0, v0g, swg, gr_h);
    // M=8208 (=513 row tiles, odd) -> MT=1 variant, grid.y = ceil(513/4)
    gemm_wmma_kernel<1, 4, 0><<<dim3(512 / 64, (513 + 3) / 4), blkG, 0, stream>>>(
        gr_h, 512, Wo_h, 512, bo, growth_out, 512, B_ * 513, 512, 512);

    // 6. h = LN(res1 - growth[:,1:])
    ln1_kernel<<<dim3(8192), blk256, 0, stream>>>(res1_32, growth_out, g1, b1ln, h_f32, h_h);

    // 7. FFN: hidden = relu(h @ W1^T) (WMMA, f16 out); ff = hidden @ W2^T (WMMA)
    gemm_wmma_kernel<2, 4, 1><<<dim3(DFF_ / 64, 8192 / 128), blkG, 0, stream>>>(
        h_h, 512, W1_h, 512, nullptr, hidden_h, DFF_, 8192, DFF_, 512);
    gemm_wmma_kernel<2, 4, 0><<<dim3(512 / 64, 8192 / 128), blkG, 0, stream>>>(
        hidden_h, DFF_, W2_h, DFF_, nullptr, v_f32 /*ff*/, 512, 8192, 512, DFF_);

    // 8. h_out = LN(h + ff)
    ln2_kernel<<<dim3(8192), blk256, 0, stream>>>(h_f32, v_f32, g2, b2ln, h_out);

    // 9. gp / sp projections (N=7), then level ES -> lvl_out
    proj7_kernel<<<dim3(224), blk256, 0, stream>>>(growth_out, 513, 1, Wg, bg, gp);
    proj7_kernel<<<dim3(224), blk256, 0, stream>>>(season_out, TP_, 0, Ws, bs, sp);
    level_es_kernel<<<dim3(1), dim3(128), 0, stream>>>(level, sp, gp, v0l, swl, lvl_out);
}
